// AttentionBlock_49228915146849
// MI455X (gfx1250) — matrix operand order repaired and checked by reading, not compiled, not run on hardware
//
#include <hip/hip_runtime.h>
#include <math.h>

// ---------------------------------------------------------------------------
// AttentionBlock: GroupNorm -> QKV (1x1 conv) -> 4-head flash attention
//                 -> proj (1x1 conv) -> +residual
// Shapes: B=4, C=256, H=W=64 (S=4096), heads=4, d=64, groups=8.
// CDNA5 strategy:
//  * v_wmma_f32_16x16x32_f16 for all three GEMM stages (f16 in, f32 accum)
//  * flash-attention tiling (no 4096x4096 score matrix)
//  * global_load_async_to_lds_b128 + ASYNCcnt double-buffering for K/V tiles
// ---------------------------------------------------------------------------

typedef __attribute__((ext_vector_type(16))) _Float16 v16h;
typedef __attribute__((ext_vector_type(2)))  _Float16 v2h;
typedef __attribute__((ext_vector_type(8)))  float    v8f;

#define BB 4
#define CC 256
#define SS 4096
#define NH 4
#define DD 64

// 16-bit A-matrix 16x32 layout (contiguous k = 16*lh + s, matching B): pair j of lane-half lh starts at:
__device__ __forceinline__ int a_kbase(int j, int lh) {
    return lh * 16 + j * 2;
}

// Load one 16x32 f16 A-fragment for this lane from a row pointer (row = M=lane&15).
__device__ __forceinline__ v16h load_a_frag(const _Float16* __restrict__ row, int lh) {
    v16h a;
#pragma unroll
    for (int j = 0; j < 8; ++j) {
        v2h p = *(const v2h*)(row + a_kbase(j, lh));
        a[2 * j]     = p.x;
        a[2 * j + 1] = p.y;
    }
    return a;
}

// Async global->LDS copy, 16 bytes per lane (one wave-wide instruction).
// Generic LDS pointers hold the DS byte offset in their low 32 bits.
__device__ __forceinline__ void async_b128(void* lds, const void* g) {
    unsigned ldsa = (unsigned)(uintptr_t)lds;
    asm volatile("global_load_async_to_lds_b128 %0, %1, off"
                 :
                 : "v"(ldsa), "v"(g)
                 : "memory");
}

// ---------------------------------------------------------------------------
// GroupNorm statistics: one block per (b, group); group slab is contiguous.
// ---------------------------------------------------------------------------
__global__ __launch_bounds__(256) void gn_stats_kernel(const float* __restrict__ x,
                                                       float* __restrict__ stats) {
    const size_t base = (size_t)blockIdx.x * (32u * SS);  // 32 ch * 4096
    const int nvec = (32 * SS) / 4;
    float s = 0.f, ss = 0.f;
    const float4* xv = (const float4*)(x + base);
    for (int i = threadIdx.x; i < nvec; i += 256) {
        float4 v = xv[i];
        s  += v.x + v.y + v.z + v.w;
        ss += v.x * v.x + v.y * v.y + v.z * v.z + v.w * v.w;
    }
#pragma unroll
    for (int off = 16; off >= 1; off >>= 1) {
        s  += __shfl_xor(s, off, 32);
        ss += __shfl_xor(ss, off, 32);
    }
    __shared__ float rs[8], rss[8];
    if ((threadIdx.x & 31) == 0) { rs[threadIdx.x >> 5] = s; rss[threadIdx.x >> 5] = ss; }
    __syncthreads();
    if (threadIdx.x == 0) {
        float S = 0.f, SSm = 0.f;
#pragma unroll
        for (int w = 0; w < 8; ++w) { S += rs[w]; SSm += rss[w]; }
        const float inv_n = 1.0f / (32.0f * (float)SS);
        float mu  = S * inv_n;
        float var = SSm * inv_n - mu * mu;
        stats[blockIdx.x * 2 + 0] = mu;
        stats[blockIdx.x * 2 + 1] = rsqrtf(var + 1e-5f);
    }
}

// Apply GN affine, emit f16 activations [b][c][s].
__global__ __launch_bounds__(256) void gn_apply_kernel(const float* __restrict__ x,
                                                       const float* __restrict__ stats,
                                                       const float* __restrict__ gw,
                                                       const float* __restrict__ gb,
                                                       _Float16* __restrict__ xn) {
    size_t i = ((size_t)blockIdx.x * 256 + threadIdx.x) * 4;
    int c  = (int)((i >> 12) & 255);     // i / 4096 % 256
    int bg = (int)(i >> 17);             // i / (32*4096) = b*8+g
    float mu = stats[bg * 2 + 0];
    float rs = stats[bg * 2 + 1];
    float w = gw[c] * rs;
    float bias = gb[c] - mu * w;
    float4 v = *(const float4*)(x + i);
    xn[i + 0] = (_Float16)(v.x * w + bias);
    xn[i + 1] = (_Float16)(v.y * w + bias);
    xn[i + 2] = (_Float16)(v.z * w + bias);
    xn[i + 3] = (_Float16)(v.w * w + bias);
}

__global__ __launch_bounds__(256) void f32_to_f16_kernel(const float* __restrict__ a,
                                                         _Float16* __restrict__ o, int n) {
    int i = blockIdx.x * 256 + threadIdx.x;
    if (i < n) o[i] = (_Float16)a[i];
}

// ---------------------------------------------------------------------------
// Generic WMMA GEMM:  out[o][n] = sum_k W[o][k] * X[k][n] + bias[o]
// Block: 8 waves, 128 o-rows x 64 n-cols.  K consumed in chunks of 32.
// mode 0: scatter f16 into Q/K (s-major, d contig) and V transposed [bh][d][s]
// mode 1: f32 out + residual (projection)
// ---------------------------------------------------------------------------
__global__ __launch_bounds__(256) void gemm_wmma_kernel(
    const _Float16* __restrict__ w16,   // [M][K]
    const _Float16* __restrict__ xin,   // [B][K][N]
    const float* __restrict__ bias,     // [M]
    int M, int K, int N, int mode,
    _Float16* __restrict__ qout, _Float16* __restrict__ kout, _Float16* __restrict__ vtout,
    const float* __restrict__ xres, float* __restrict__ outf) {
    __shared__ __align__(32) _Float16 lx[64][48];  // [n][k] transposed, stride 96B

    const int b = blockIdx.z;
    const int nbase = blockIdx.x * 64;
    const int oblock = blockIdx.y * 128;
    const int tid = threadIdx.x;
    const int wave = tid >> 5, lane = tid & 31;
    const int m16 = lane & 15, lh = lane >> 4;
    const _Float16* xb = xin + (size_t)b * K * N;

    v8f acc[4] = {};

    for (int kc = 0; kc < K; kc += 32) {
        // cooperative transpose-load of the 32(k) x 64(n) activation tile
        {
            int k = tid >> 3;
            int n0 = (tid & 7) * 8;
            union { uint4 u; _Float16 h[8]; } tmp;
            tmp.u = *(const uint4*)(xb + (size_t)(kc + k) * N + nbase + n0);
#pragma unroll
            for (int jj = 0; jj < 8; ++jj) lx[n0 + jj][k] = tmp.h[jj];
        }
        __syncthreads();

        const int orow = oblock + wave * 16 + m16;
        v16h a = load_a_frag(w16 + (size_t)orow * K + kc, lh);
#pragma unroll
        for (int t = 0; t < 4; ++t) {
            v16h bf = *(const v16h*)&lx[t * 16 + m16][lh * 16];
            acc[t] = __builtin_amdgcn_wmma_f32_16x16x32_f16(
                false, a, false, bf, (short)0, acc[t], false, false);
        }
        __syncthreads();
    }

    // C/D layout: row M = r + 8*lh, col N = lane&15 within each 16-wide n-tile
#pragma unroll
    for (int t = 0; t < 4; ++t) {
#pragma unroll
        for (int r = 0; r < 8; ++r) {
            int o = oblock + wave * 16 + r + 8 * lh;
            int n = nbase + t * 16 + m16;
            float v = acc[t][r] + bias[o];
            if (mode == 0) {
                if (o < 256) {
                    int h = o >> 6, d = o & 63;
                    qout[(((size_t)b * NH + h) * N + n) * DD + d] = (_Float16)v;
                } else if (o < 512) {
                    int o2 = o - 256, h = o2 >> 6, d = o2 & 63;
                    kout[(((size_t)b * NH + h) * N + n) * DD + d] = (_Float16)v;
                } else {
                    int o2 = o - 512, h = o2 >> 6, d = o2 & 63;
                    vtout[(((size_t)b * NH + h) * DD + d) * N + n] = (_Float16)v;
                }
            } else {
                size_t idx = ((size_t)b * M + o) * (size_t)N + n;
                outf[idx] = v + xres[idx];
            }
        }
    }
}

// ---------------------------------------------------------------------------
// Flash attention: block = 128 query rows (8 waves x 16), streams K/V in
// 64-wide tiles through LDS with double-buffered async global->LDS copies.
// Online softmax in registers, all GEMMs WMMA.
// ---------------------------------------------------------------------------
__global__ __launch_bounds__(256) void flash_attn_kernel(
    const _Float16* __restrict__ Q,    // [bh][s][d]
    const _Float16* __restrict__ Kt,   // [bh][s][d]
    const _Float16* __restrict__ Vt,   // [bh][d][s]
    _Float16* __restrict__ att) {      // [b][c=256][s]
    __shared__ __align__(32) _Float16 lk[2][64][80];    // K tiles, rows=s_local
    __shared__ __align__(32) _Float16 lv[2][64][80];    // V tiles, rows=d
    __shared__ __align__(32) _Float16 lp[8][16][72];    // per-wave P staging

    const int bh = blockIdx.y;       // b*4 + h
    const int qtile = blockIdx.x;    // 0..31 (128 rows each)
    const int tid = threadIdx.x;
    const int wave = tid >> 5, lane = tid & 31;
    const int m16 = lane & 15, lh = lane >> 4;
    const int qrow = qtile * 128 + wave * 16 + m16;

    // async tile fetch: 4 wave-wide b128 copies per tile (2 for K, 2 for V)
    const int cr  = tid >> 2;            // 0..63: K row / V d-row
    const int seg = (tid & 3) * 16;      // 0,16,32,48 halves
    auto issue_tile = [&](int kt, int p) {
        const _Float16* ks = Kt + ((size_t)bh * SS + kt * 64 + cr) * DD + seg;
        async_b128(&lk[p][cr][seg],     ks);
        async_b128(&lk[p][cr][seg + 8], ks + 8);
        const _Float16* vs = Vt + ((size_t)bh * DD + cr) * SS + kt * 64 + seg;
        async_b128(&lv[p][cr][seg],     vs);
        async_b128(&lv[p][cr][seg + 8], vs + 8);
    };

    // Q A-fragments with softmax scale (1/sqrt(64)) folded in
    v16h aq0, aq1;
    {
        const _Float16* qp = Q + ((size_t)bh * SS + qrow) * DD;
        aq0 = load_a_frag(qp, lh);
        aq1 = load_a_frag(qp + 32, lh);
        const _Float16 sc = (_Float16)0.125f;
#pragma unroll
        for (int i = 0; i < 16; ++i) { aq0[i] *= sc; aq1[i] *= sc; }
    }

    v8f acc[4] = {};
    float mI[8], lI[8];
#pragma unroll
    for (int r = 0; r < 8; ++r) { mI[r] = -INFINITY; lI[r] = 0.0f; }

    issue_tile(0, 0);  // prime the pipeline

    const int NT = SS / 64;
    for (int kt = 0; kt < NT; ++kt) {
        const int p = kt & 1;
        if (kt + 1 < NT) {
            issue_tile(kt + 1, p ^ 1);  // prefetch next tile into other buffer
            // async loads complete in order: <=4 outstanding => tile kt landed
            asm volatile("s_wait_asynccnt 0x4" ::: "memory");
        } else {
            asm volatile("s_wait_asynccnt 0x0" ::: "memory");
        }
        __syncthreads();

        // S = (Q*scale) @ K^T   (16 x 64 per wave, 4 n-tiles x 2 k-steps)
        v8f sacc[4];
#pragma unroll
        for (int t = 0; t < 4; ++t) {
            v8f s = {};
            v16h b0 = *(const v16h*)&lk[p][t * 16 + m16][lh * 16];
            s = __builtin_amdgcn_wmma_f32_16x16x32_f16(false, aq0, false, b0, (short)0, s, false, false);
            v16h b1 = *(const v16h*)&lk[p][t * 16 + m16][32 + lh * 16];
            s = __builtin_amdgcn_wmma_f32_16x16x32_f16(false, aq1, false, b1, (short)0, s, false, false);
            sacc[t] = s;
        }

        // online softmax: rows live on (vgpr slot r, lane-half), cols on 16 lanes
        float pr[4][8];
#pragma unroll
        for (int r = 0; r < 8; ++r) {
            float mx = fmaxf(fmaxf(sacc[0][r], sacc[1][r]), fmaxf(sacc[2][r], sacc[3][r]));
#pragma unroll
            for (int off = 1; off < 16; off <<= 1) mx = fmaxf(mx, __shfl_xor(mx, off, 32));
            float mnew  = fmaxf(mI[r], mx);
            float alpha = __expf(mI[r] - mnew);
            float rsum = 0.f;
#pragma unroll
            for (int t = 0; t < 4; ++t) {
                float ptv = __expf(sacc[t][r] - mnew);
                pr[t][r] = ptv;
                rsum += ptv;
            }
#pragma unroll
            for (int off = 1; off < 16; off <<= 1) rsum += __shfl_xor(rsum, off, 32);
            lI[r] = lI[r] * alpha + rsum;
            mI[r] = mnew;
#pragma unroll
            for (int t = 0; t < 4; ++t) acc[t][r] *= alpha;
        }

        // C-layout -> A-layout conversion through per-wave LDS staging
#pragma unroll
        for (int t = 0; t < 4; ++t)
#pragma unroll
            for (int r = 0; r < 8; ++r)
                lp[wave][r + 8 * lh][t * 16 + m16] = (_Float16)pr[t][r];
        // same-wave LDS ops are in-order: safe to read back without a barrier
        v16h ap0 = load_a_frag(&lp[wave][m16][0], lh);
        v16h ap1 = load_a_frag(&lp[wave][m16][32], lh);

        // O += P @ V   (contraction over the 64 tile columns, 2 k-steps)
#pragma unroll
        for (int t = 0; t < 4; ++t) {
            v16h b0 = *(const v16h*)&lv[p][t * 16 + m16][lh * 16];
            acc[t] = __builtin_amdgcn_wmma_f32_16x16x32_f16(false, ap0, false, b0, (short)0, acc[t], false, false);
            v16h b1 = *(const v16h*)&lv[p][t * 16 + m16][32 + lh * 16];
            acc[t] = __builtin_amdgcn_wmma_f32_16x16x32_f16(false, ap1, false, b1, (short)0, acc[t], false, false);
        }
        __syncthreads();  // all reads of buffer p done before it is refilled
    }

    // finalize: divide by row sums, store att[b][h*64+d][s] (f16)
#pragma unroll
    for (int r = 0; r < 8; ++r) {
        float inv = 1.0f / lI[r];
#pragma unroll
        for (int t = 0; t < 4; ++t) acc[t][r] *= inv;
    }
    const int b = bh >> 2, h = bh & 3;
#pragma unroll
    for (int t = 0; t < 4; ++t) {
        int d = t * 16 + m16;
#pragma unroll
        for (int r = 0; r < 8; ++r) {
            int srow = qtile * 128 + wave * 16 + r + 8 * lh;
            att[((size_t)b * CC + h * DD + d) * (size_t)SS + srow] = (_Float16)acc[t][r];
        }
    }
}

// ---------------------------------------------------------------------------
extern "C" void kernel_launch(void* const* d_in, const int* in_sizes, int n_in,
                              void* d_out, int out_size, void* d_ws, size_t ws_size,
                              hipStream_t stream) {
    const float* x      = (const float*)d_in[0];
    const float* gn_w   = (const float*)d_in[1];
    const float* gn_b   = (const float*)d_in[2];
    const float* qkv_w  = (const float*)d_in[3];
    const float* qkv_b  = (const float*)d_in[4];
    const float* proj_w = (const float*)d_in[5];
    const float* proj_b = (const float*)d_in[6];
    float* out = (float*)d_out;

    const size_t ACT = (size_t)BB * CC * SS;  // 4,194,304 elements

    char* w = (char*)d_ws;
    float*     stats = (float*)w;         w += 1024;
    _Float16*  xn    = (_Float16*)w;      w += ACT * 2;
    _Float16*  wq16  = (_Float16*)w;      w += (size_t)768 * 256 * 2;
    _Float16*  wp16  = (_Float16*)w;      w += (size_t)256 * 256 * 2;
    _Float16*  Qh    = (_Float16*)w;      w += ACT * 2;
    _Float16*  Kh    = (_Float16*)w;      w += ACT * 2;
    _Float16*  Vth   = (_Float16*)w;      w += ACT * 2;
    _Float16*  att   = (_Float16*)w;      w += ACT * 2;
    (void)ws_size; (void)in_sizes; (void)n_in; (void)out_size;

    // 1) GroupNorm stats + apply (f16 activations)
    gn_stats_kernel<<<BB * 8, 256, 0, stream>>>(x, stats);
    gn_apply_kernel<<<(unsigned)(ACT / 1024), 256, 0, stream>>>(x, stats, gn_w, gn_b, xn);

    // 2) weight conversion to f16
    f32_to_f16_kernel<<<768, 256, 0, stream>>>(qkv_w, wq16, 768 * 256);
    f32_to_f16_kernel<<<256, 256, 0, stream>>>(proj_w, wp16, 256 * 256);

    // 3) QKV GEMM (M=768, K=256, N=4096), scatter to Q/K/V^T
    gemm_wmma_kernel<<<dim3(SS / 64, 768 / 128, BB), 256, 0, stream>>>(
        wq16, xn, qkv_b, 768, 256, SS, 0, Qh, Kh, Vth, nullptr, nullptr);

    // 4) flash attention
    flash_attn_kernel<<<dim3(SS / 128, BB * NH), 256, 0, stream>>>(Qh, Kh, Vth, att);

    // 5) projection GEMM + bias + residual -> f32 output
    gemm_wmma_kernel<<<dim3(SS / 64, 256 / 128, BB), 256, 0, stream>>>(
        wp16, att, proj_b, 256, 256, SS, 1, nullptr, nullptr, nullptr, x, out);
}
